// PyramidCompressor_1211180777490
// MI455X (gfx1250) — matrix operand order repaired and checked by reading, not compiled, not run on hardware
//
#include <hip/hip_runtime.h>

// ---------------------------------------------------------------------------
// PyramidCompressor for MI455X (gfx1250, wave32, WMMA).
//
// Effective reference math (level-0 re-capture quirk => levels are only 1|2):
//   h = relu(LN(x @ We0 + be0));  h = relu(LN(h @ We1 + be1))
//   if level==2: h = LN( relu(LN(h @ We2 + be2)) @ Wd2 + bd2 )
//   h = LN(h @ Wd1 + bd1);  h = LN(h @ Wd0 + bd0);  out = x + h
//
// Strategy: bf16 WMMA (v_wmma_f32_16x16x32_bf16) with f32 accumulate.
// Weights pre-packed to bf16 in WMMA B-operand lane layout in the workspace
// (zero padded: 204->224, 65->96). One workgroup (8 waves) per 16-token tile;
// activations ping-pong LDS f32 <-> LDS bf16.
// ---------------------------------------------------------------------------

typedef __attribute__((ext_vector_type(16))) __bf16 v16bf;
typedef __attribute__((ext_vector_type(8)))  float  v8f;

#define D0   1024
#define D1   512
#define D2T  204
#define D2P  224
#define D3T  65
#define D3P  96
#define LN_EPS 1e-5f

union BF16Frag { v16bf v; uint4 q[2]; };

__device__ __forceinline__ unsigned short f2bf(float f) {
  union { float f; unsigned u; } v; v.f = f;
  unsigned r = v.u + 0x7FFFu + ((v.u >> 16) & 1u);   // round-to-nearest-even
  return (unsigned short)(r >> 16);
}

// ---------------------------------------------------------------------------
// Weight packer: f32 (Ktrue x Ntrue row-major) -> bf16 WMMA-B tile layout.
// Packed element index: ((t*kblocks + kb)*32 + lane)*16 + p
//   lane<16 : col = t*16 + lane,    K positions p -> k = kb*32 + p
//   lane>=16: col = t*16 + lane-16, K positions p -> k = kb*32 + 16 + p
// Out-of-range (K or N padding) -> 0.
// ---------------------------------------------------------------------------
__global__ void pack_weight(const float* __restrict__ src,
                            unsigned short* __restrict__ dst,
                            int Ktrue, int Ntrue, int kblocks, int ntiles) {
  int idx = blockIdx.x * blockDim.x + threadIdx.x;
  int total = kblocks * ntiles * 512;
  if (idx >= total) return;
  int p    = idx & 15;
  int lane = (idx >> 4) & 31;
  int rest = idx >> 9;
  int kb   = rest % kblocks;
  int t    = rest / kblocks;
  int col  = t * 16 + (lane & 15);
  int k    = kb * 32 + (lane >> 4) * 16 + p;
  float v = 0.0f;
  if (k < Ktrue && col < Ntrue) v = src[(size_t)k * (size_t)Ntrue + col];
  dst[idx] = f2bf(v);
}

struct KParams {
  const float* keys;
  const float* values;
  const float* imp;
  const unsigned short *We0, *We1, *We2, *Wd2, *Wd1, *Wd0;
  const float *be0, *be1, *be2, *bd2, *bd1, *bd0;
  const float *ge0, *ge1, *ge2, *gd2, *gd1, *gd0;
  const float *ce0, *ce1, *ce2, *cd2, *cd1, *cd0;
  float* out;
  int tokens;   // B*S
};

// ---------------------------------------------------------------------------
// GEMM: H[16 x Npad] = bf16(A[16 x K]) @ Wpack + bias.
// A in LDS, row-major ushort, ldA elems. Wpack global, pre-swizzled bf16.
// 8 waves split the Ntiles; each wave's lane mapping (A/B use contiguous K, not the ISA table order):
//   A (16x32 bf16):  lane&15 = M row; lane>>4 selects K half
//       positions 0..15 -> K = kb*32 + half*16 + p
//   C/D (16x16 f32): vgpr r: lanes 0-15 -> (M=r, N=lane); 16-31 -> (M=r+8)
// ---------------------------------------------------------------------------
__device__ __forceinline__ void gemm_tile(const unsigned short* __restrict__ Abuf,
                                          int ldA, int K,
                                          const unsigned short* __restrict__ Wp,
                                          const float* __restrict__ bias,
                                          int Ntrue, int Ntiles,
                                          float* __restrict__ Hbuf, int ldH,
                                          int wave, int lane) {
  const int half = lane >> 4;
  const int nlo  = lane & 15;
  const int kblocks = K >> 5;
  const int arow = nlo;                     // A: lane&15 = matrix row M
  for (int t = wave; t < Ntiles; t += 8) {
    v8f acc = {};
    const unsigned short* wb = Wp + ((size_t)(t * kblocks) << 9) + (lane << 4);
    for (int kb = 0; kb < kblocks; ++kb) {
      BF16Frag a, b;
      const unsigned short* ar = Abuf + arow * ldA + kb * 32 + half * 16;
      a.q[0] = *reinterpret_cast<const uint4*>(ar);
      a.q[1] = *reinterpret_cast<const uint4*>(ar + 8);
      const uint4* bp = reinterpret_cast<const uint4*>(wb + ((size_t)kb << 9));
      b.q[0] = bp[0];
      b.q[1] = bp[1];
      acc = __builtin_amdgcn_wmma_f32_16x16x32_bf16(
                false, a.v, false, b.v, (short)0, acc, false, false);
    }
    const int col = t * 16 + nlo;
    const float bv = (col < Ntrue) ? bias[col] : 0.0f;
    const int row0 = half * 8;
#pragma unroll
    for (int r = 0; r < 8; ++r)
      Hbuf[(row0 + r) * ldH + col] = acc[r] + bv;
  }
}

// ---------------------------------------------------------------------------
// LayerNorm over Dtrue of H[16 x ldH], optional ReLU, write bf16 A[16 x ldA]
// (zero-filling pad cols), optionally save f32 copy (Fsave, stride ldA),
// optionally per-row select: rows with lev[row]!=2 take Hsel instead of LN(y).
// 16 threads per row (half-wave); shfl-xor reduction stays inside the half.
// ---------------------------------------------------------------------------
__device__ __forceinline__ void ln_stage(const float* __restrict__ H, int ldH,
                                         const float* __restrict__ g,
                                         const float* __restrict__ c,
                                         int Dtrue, int Dpad,
                                         unsigned short* __restrict__ Aout, int ldA,
                                         bool relu,
                                         float* __restrict__ Fsave,
                                         const int* __restrict__ lev,
                                         const float* __restrict__ Hsel,
                                         int tid) {
  const int row = tid >> 4;
  const int sub = tid & 15;
  const float* hr = H + row * ldH;
  float sum = 0.0f, ss = 0.0f;
  for (int col = sub; col < Dtrue; col += 16) {
    float x = hr[col];
    sum += x; ss += x * x;
  }
#pragma unroll
  for (int m = 8; m >= 1; m >>= 1) {
    sum += __shfl_xor(sum, m, 32);
    ss  += __shfl_xor(ss,  m, 32);
  }
  const float inv  = 1.0f / (float)Dtrue;
  const float mu   = sum * inv;
  const float var  = ss * inv - mu * mu;
  const float rstd = rsqrtf(var + LN_EPS);
  const bool takeLN = lev ? (lev[row] == 2) : true;
  for (int col = sub; col < Dpad; col += 16) {
    float y = 0.0f;
    if (col < Dtrue) {
      y = (hr[col] - mu) * rstd * g[col] + c[col];
      if (relu) y = fmaxf(y, 0.0f);
    }
    if (lev && !takeLN) y = Hsel[row * ldA + col];
    Aout[row * ldA + col] = f2bf(y);
    if (Fsave) Fsave[row * ldA + col] = y;
  }
}

__global__ __launch_bounds__(256)
void pyramid_kernel(KParams P) {
  __shared__ __align__(16) unsigned short Abuf[16 * D0];   // 32 KB bf16 activations
  __shared__ __align__(16) float          Hbuf[16 * D0];   // 64 KB f32 GEMM out
  __shared__ __align__(16) float          H1b [16 * D2P];  // 14 KB saved enc1 act
  __shared__ int slev[16];

  const int tile   = blockIdx.x;
  const int tensor = blockIdx.y;
  const float* xin = tensor ? P.values : P.keys;
  float* outp = P.out + (size_t)tensor * (size_t)P.tokens * D0;
  const int tok0 = tile * 16;
  const int tid  = threadIdx.x;
  const int lane = tid & 31;
  const int wave = tid >> 5;

  // Level per token: quirk-folded assignment -> imp>=0.5 ? 1 : 2
  if (tid < 16)
    slev[tid] = (P.imp[tok0 + tid] >= 0.5f) ? 1 : 2;

  // Stage input x -> bf16 A (coalesced: layout matches global row-major)
  for (int idx = tid; idx < 16 * D0; idx += 256) {
    int r = idx >> 10, col = idx & (D0 - 1);
    Abuf[idx] = f2bf(xin[(size_t)(tok0 + r) * D0 + col]);
  }
  __syncthreads();

  bool hasL2 = false;
#pragma unroll
  for (int r = 0; r < 16; ++r) hasL2 |= (slev[r] == 2);

  // enc0: 1024 -> 512
  gemm_tile(Abuf, D0, D0, P.We0, P.be0, D1, D1 / 16, Hbuf, D1, wave, lane);
  __syncthreads();
  ln_stage(Hbuf, D1, P.ge0, P.ce0, D1, D1, Abuf, D1, true, nullptr, nullptr, nullptr, tid);
  __syncthreads();

  // enc1: 512 -> 204 (pad 224); keep f32 copy for level-1 bypass
  gemm_tile(Abuf, D1, D1, P.We1, P.be1, D2T, D2P / 16, Hbuf, D2P, wave, lane);
  __syncthreads();
  ln_stage(Hbuf, D2P, P.ge1, P.ce1, D2T, D2P, Abuf, D2P, true, H1b, nullptr, nullptr, tid);
  __syncthreads();

  if (hasL2) {
    // enc2: 204 -> 65 (pad 96)
    gemm_tile(Abuf, D2P, D2P, P.We2, P.be2, D3T, D3P / 16, Hbuf, D3P, wave, lane);
    __syncthreads();
    ln_stage(Hbuf, D3P, P.ge2, P.ce2, D3T, D3P, Abuf, D3P, true, nullptr, nullptr, nullptr, tid);
    __syncthreads();
    // dec2: 65 -> 204, then per-row select vs saved enc1 activation
    gemm_tile(Abuf, D3P, D3P, P.Wd2, P.bd2, D2T, D2P / 16, Hbuf, D2P, wave, lane);
    __syncthreads();
    ln_stage(Hbuf, D2P, P.gd2, P.cd2, D2T, D2P, Abuf, D2P, false, nullptr, slev, H1b, tid);
    __syncthreads();
  }

  // dec1: 204 -> 512
  gemm_tile(Abuf, D2P, D2P, P.Wd1, P.bd1, D1, D1 / 16, Hbuf, D1, wave, lane);
  __syncthreads();
  ln_stage(Hbuf, D1, P.gd1, P.cd1, D1, D1, Abuf, D1, false, nullptr, nullptr, nullptr, tid);
  __syncthreads();

  // dec0: 512 -> 1024
  gemm_tile(Abuf, D1, D1, P.Wd0, P.bd0, D0, D0 / 16, Hbuf, D0, wave, lane);
  __syncthreads();

  // Final LN + residual, fused global store
  {
    const int row = tid >> 4;
    const int sub = tid & 15;
    const float* hr = Hbuf + row * D0;
    float sum = 0.0f, ss = 0.0f;
    for (int col = sub; col < D0; col += 16) {
      float x = hr[col];
      sum += x; ss += x * x;
    }
#pragma unroll
    for (int m = 8; m >= 1; m >>= 1) {
      sum += __shfl_xor(sum, m, 32);
      ss  += __shfl_xor(ss,  m, 32);
    }
    const float mu   = sum * (1.0f / D0);
    const float var  = ss * (1.0f / D0) - mu * mu;
    const float rstd = rsqrtf(var + LN_EPS);
    const float* xr = xin + (size_t)(tok0 + row) * D0;
    float* orow = outp + (size_t)(tok0 + row) * D0;
    for (int col = sub; col < D0; col += 16) {
      float y = (hr[col] - mu) * rstd * P.gd0[col] + P.cd0[col];
      orow[col] = xr[col] + y;
    }
  }
}

// ---------------------------------------------------------------------------
// Workspace layout (bf16 elems): packed weights, all sizes multiple of 512.
// ---------------------------------------------------------------------------
#define O_WE0 0u
#define O_WE1 524288u     // 32*32*512
#define O_WE2 638976u     // + 16*14*512
#define O_WD2 660480u     // + 7*6*512
#define O_WD1 681984u     // + 3*14*512
#define O_WD0 796672u     // + 7*32*512 ; end = 1320960 elems (~2.52 MB)

extern "C" void kernel_launch(void* const* d_in, const int* in_sizes, int n_in,
                              void* d_out, int out_size, void* d_ws, size_t ws_size,
                              hipStream_t stream) {
  const float* keys   = (const float*)d_in[0];
  const float* values = (const float*)d_in[1];
  const float* imp    = (const float*)d_in[2];
  // params flatten: per level {We, be, ge, ce, Wd, bd, gd, cd}
  const float* prm[3][8];
  for (int l = 0; l < 3; ++l)
    for (int j = 0; j < 8; ++j)
      prm[l][j] = (const float*)d_in[3 + l * 8 + j];

  unsigned short* ws = (unsigned short*)d_ws;
  unsigned short* We0 = ws + O_WE0;
  unsigned short* We1 = ws + O_WE1;
  unsigned short* We2 = ws + O_WE2;
  unsigned short* Wd2 = ws + O_WD2;
  unsigned short* Wd1 = ws + O_WD1;
  unsigned short* Wd0 = ws + O_WD0;

  struct PackJob { const float* src; unsigned short* dst; int Kt, Nt, kb, nt; };
  const PackJob jobs[6] = {
    { prm[0][0], We0, D0,  D1,  D0 / 32,  D1 / 16 },
    { prm[1][0], We1, D1,  D2T, D1 / 32,  D2P / 16 },
    { prm[2][0], We2, D2T, D3T, D2P / 32, D3P / 16 },
    { prm[2][4], Wd2, D3T, D2T, D3P / 32, D2P / 16 },
    { prm[1][4], Wd1, D2T, D1,  D2P / 32, D1 / 16 },
    { prm[0][4], Wd0, D1,  D0,  D1 / 32,  D0 / 16 },
  };
  for (int j = 0; j < 6; ++j) {
    int total = jobs[j].kb * jobs[j].nt * 512;
    pack_weight<<<(total + 255) / 256, 256, 0, stream>>>(
        jobs[j].src, jobs[j].dst, jobs[j].Kt, jobs[j].Nt, jobs[j].kb, jobs[j].nt);
  }

  KParams P;
  P.keys = keys; P.values = values; P.imp = imp;
  P.We0 = We0; P.We1 = We1; P.We2 = We2; P.Wd2 = Wd2; P.Wd1 = Wd1; P.Wd0 = Wd0;
  P.be0 = prm[0][1]; P.be1 = prm[1][1]; P.be2 = prm[2][1];
  P.bd2 = prm[2][5]; P.bd1 = prm[1][5]; P.bd0 = prm[0][5];
  P.ge0 = prm[0][2]; P.ge1 = prm[1][2]; P.ge2 = prm[2][2];
  P.gd2 = prm[2][6]; P.gd1 = prm[1][6]; P.gd0 = prm[0][6];
  P.ce0 = prm[0][3]; P.ce1 = prm[1][3]; P.ce2 = prm[2][3];
  P.cd2 = prm[2][7]; P.cd1 = prm[1][7]; P.cd0 = prm[0][7];
  P.out = (float*)d_out;
  P.tokens = in_sizes[2];                       // B*S = 32768

  const int ntile = P.tokens / 16;              // 2048 tiles per tensor
  pyramid_kernel<<<dim3(ntile, 2), dim3(256), 0, stream>>>(P);
}
